// BiologicalGAT_73340861546586
// MI455X (gfx1250) — matrix operand order repaired and checked by reading, not compiled, not run on hardware
//
#include <hip/hip_runtime.h>
#include <hip/hip_bf16.h>

#define NN    10000
#define EE    160000
#define ET    170000     // edges + self loops
#define HID   512
#define NG    8

typedef __attribute__((ext_vector_type(16))) _Float16 v16h;
typedef __attribute__((ext_vector_type(8)))  _Float16 v8h;
typedef __attribute__((ext_vector_type(8)))  float    v8f;

// ---------- helpers ----------
__device__ inline void edge_sd(const int* ei, int e, int& s, int& d) {
    if (e < EE) { s = ei[e]; d = ei[EE + e]; }
    else        { s = e - EE; d = e - EE; }
}
__device__ inline unsigned f2mono(float f) {
    unsigned u = __float_as_uint(f);
    return (u >> 31) ? ~u : (u | 0x80000000u);
}
__device__ inline float mono2f(unsigned m) {
    unsigned u = (m >> 31) ? (m & 0x7FFFFFFFu) : ~m;
    return __uint_as_float(u);
}

// ---------- kernels ----------
__global__ void k_zero(float* p, int n) {
    int i = blockIdx.x * blockDim.x + threadIdx.x;
    if (i < n) p[i] = 0.0f;
}

// W [K=512][N=512] f32 row-major -> Wt [N][K] f16
__global__ void k_wt(const float* __restrict__ W, _Float16* __restrict__ Wt) {
    int i = blockIdx.x * blockDim.x + threadIdx.x;
    if (i < 512 * 512) {
        int k = i >> 9, n = i & 511;
        Wt[n * 512 + k] = (_Float16)W[k * 512 + n];
    }
}

__global__ void k_addemb(const float* __restrict__ x, const int* __restrict__ nt,
                         const float* __restrict__ emb, float* __restrict__ xo) {
    int i = blockIdx.x * blockDim.x + threadIdx.x;
    if (i < NN * HID) {
        int n = i >> 9, c = i & 511;
        xo[i] = x[i] + emb[nt[n] * HID + c];
    }
}

// H = X(f32 [10000][512]) @ W  using Wt(f16 [N][K]); each wave: 16(M)x128(N) tile
__global__ __launch_bounds__(256) void k_gemm(const float* __restrict__ X,
                                              const _Float16* __restrict__ Wt,
                                              float* __restrict__ H) {
    const int lane  = threadIdx.x & 31;
    const int w     = threadIdx.x >> 5;
    const int mt    = blockIdx.x * 2 + (w >> 2);
    if (mt >= 625) return;                 // 10000/16 = 625 M-tiles
    const int m0    = mt * 16;
    const int n0    = (w & 3) * 128;
    const int lh    = lane >> 4;           // half-wave selector
    const int lm    = lane & 15;

    const float*    ap = X + (m0 + lm) * 512 + lh * 8;   // A row, per-lane K base
    const _Float16* bp = Wt + lh * 8;                    // + n*512 + k0

    v8f c[8];
    #pragma unroll
    for (int j = 0; j < 8; j++)
        #pragma unroll
        for (int r = 0; r < 8; r++) c[j][r] = 0.0f;

    for (int k0 = 0; k0 < 512; k0 += 32) {
        // A fragment: K [base..base+7] and [base+16..base+23], converted f32->f16
        float4 f0 = *(const float4*)(ap + k0);
        float4 f1 = *(const float4*)(ap + k0 + 4);
        float4 f2 = *(const float4*)(ap + k0 + 16);
        float4 f3 = *(const float4*)(ap + k0 + 20);
        v16h a;
        a[0]=(_Float16)f0.x; a[1]=(_Float16)f0.y; a[2]=(_Float16)f0.z; a[3]=(_Float16)f0.w;
        a[4]=(_Float16)f1.x; a[5]=(_Float16)f1.y; a[6]=(_Float16)f1.z; a[7]=(_Float16)f1.w;
        a[8]=(_Float16)f2.x; a[9]=(_Float16)f2.y; a[10]=(_Float16)f2.z; a[11]=(_Float16)f2.w;
        a[12]=(_Float16)f3.x; a[13]=(_Float16)f3.y; a[14]=(_Float16)f3.z; a[15]=(_Float16)f3.w;

        #pragma unroll
        for (int j = 0; j < 8; j++) {
            const _Float16* bpj = bp + (n0 + j * 16 + lm) * 512 + k0;
            v8h b0 = *(const v8h*)(bpj);
            v8h b1 = *(const v8h*)(bpj + 16);
            v16h b;
            #pragma unroll
            for (int i = 0; i < 8; i++) { b[i] = b0[i]; b[8 + i] = b1[i]; }
            c[j] = __builtin_amdgcn_wmma_f32_16x16x32_f16(
                false, a, false, b, (short)0, c[j], false, false);
        }
    }
    #pragma unroll
    for (int j = 0; j < 8; j++) {
        float* op = H + n0 + j * 16 + lm;
        #pragma unroll
        for (int r = 0; r < 8; r++) op[(m0 + r + 8 * lh) * 512] = c[j][r];
    }
}

// per-node attention logits: alpha_s[n,h] = sum_c H[n,h*C+c]*a_src[h,c] (flat idx == c)
__global__ __launch_bounds__(256) void k_alpha(const float* __restrict__ Hm,
                                               const float* __restrict__ asrc,
                                               const float* __restrict__ adst,
                                               float* __restrict__ als, float* __restrict__ ald,
                                               int Hh, int cshift) {
    int node = blockIdx.x * 8 + (threadIdx.x >> 5);
    if (node >= NN) return;
    int lane = threadIdx.x & 31;
    float sp[4] = {0,0,0,0}, dp[4] = {0,0,0,0};
    const float* row = Hm + node * HID;
    for (int c = lane; c < HID; c += 32) {
        float v = row[c];
        int h = c >> cshift;
        sp[h] += v * asrc[c];
        dp[h] += v * adst[c];
    }
    for (int h = 0; h < Hh; h++) {
        float a = sp[h], b = dp[h];
        for (int o = 16; o > 0; o >>= 1) { a += __shfl_xor(a, o, 32); b += __shfl_xor(b, o, 32); }
        if (lane == 0) { als[node * Hh + h] = a; ald[node * Hh + h] = b; }
    }
}

// e = leaky_relu(as[src]+ad[dst]); segment max via monotonic-uint atomicMax
__global__ void k_edge1(const int* __restrict__ ei, const float* __restrict__ als,
                        const float* __restrict__ ald, float* __restrict__ eraw,
                        unsigned* __restrict__ mx, int Hh) {
    int idx = blockIdx.x * blockDim.x + threadIdx.x;
    if (idx >= ET * Hh) return;
    int e = idx / Hh, h = idx - e * Hh;
    int s, d; edge_sd(ei, e, s, d);
    float v = als[s * Hh + h] + ald[d * Hh + h];
    v = (v > 0.0f) ? v : 0.2f * v;
    eraw[idx] = v;
    atomicMax(&mx[d * Hh + h], f2mono(v));
}

// w = exp(e - max[dst]); segment sum
__global__ void k_edge2(const int* __restrict__ ei, const unsigned* __restrict__ mx,
                        float* __restrict__ ew, float* __restrict__ sden, int Hh) {
    int idx = blockIdx.x * blockDim.x + threadIdx.x;
    if (idx >= ET * Hh) return;
    int e = idx / Hh, h = idx - e * Hh;
    int s, d; edge_sd(ei, e, s, d);
    float m  = mono2f(mx[d * Hh + h]);
    float wv = expf(ew[idx] - m);          // in-place: eraw -> w
    ew[idx] = wv;
    atomicAdd(&sden[d * Hh + h], wv);
}

// out[dst,c] += H[src,c] * w/(s+eps); one thread per (edge, 4 columns)
__global__ void k_edge3(const int* __restrict__ ei, const float* __restrict__ Hm,
                        const float* __restrict__ ew, const float* __restrict__ sden,
                        float* __restrict__ out, int Hh, int cshift) {
    int idx = blockIdx.x * blockDim.x + threadIdx.x;   // ET*128 = 21,760,000
    int e  = idx >> 7;
    if (e >= ET) return;
    int c4 = (idx & 127) << 2;
    int s, d; edge_sd(ei, e, s, d);
    int h = c4 >> cshift;
    float alpha = ew[e * Hh + h] / (sden[d * Hh + h] + 1e-16f);
    const float4 hv = *(const float4*)(Hm + s * 512 + c4);
    float* op = out + d * 512 + c4;
    atomicAdd(op + 0, hv.x * alpha);
    atomicAdd(op + 1, hv.y * alpha);
    atomicAdd(op + 2, hv.z * alpha);
    atomicAdd(op + 3, hv.w * alpha);
}

// bias + (relu) + layernorm + optional residual; one block (256 thr) per node
__global__ __launch_bounds__(256) void k_post(const float* __restrict__ out,
                                              const float* __restrict__ bias,
                                              const float* __restrict__ g,
                                              const float* __restrict__ b,
                                              const float* __restrict__ res,
                                              float* __restrict__ xn, int do_relu) {
    __shared__ float sred[8];
    int n = blockIdx.x, t = threadIdx.x;
    const float* row = out + n * 512;
    float v0 = row[t] + bias[t];
    float v1 = row[t + 256] + bias[t + 256];
    if (do_relu) { v0 = fmaxf(v0, 0.0f); v1 = fmaxf(v1, 0.0f); }

    float s = v0 + v1;
    for (int o = 16; o > 0; o >>= 1) s += __shfl_xor(s, o, 32);
    if ((t & 31) == 0) sred[t >> 5] = s;
    __syncthreads();
    float acc = 0.0f;
    for (int i = 0; i < 8; i++) acc += sred[i];
    float mu = acc * (1.0f / 512.0f);
    __syncthreads();

    float d0 = v0 - mu, d1 = v1 - mu;
    float q = d0 * d0 + d1 * d1;
    for (int o = 16; o > 0; o >>= 1) q += __shfl_xor(q, o, 32);
    if ((t & 31) == 0) sred[t >> 5] = q;
    __syncthreads();
    acc = 0.0f;
    for (int i = 0; i < 8; i++) acc += sred[i];
    float inv = rsqrtf(acc * (1.0f / 512.0f) + 1e-5f);

    float r0 = res ? res[n * 512 + t]       : 0.0f;
    float r1 = res ? res[n * 512 + t + 256] : 0.0f;
    xn[n * 512 + t]       = d0 * inv * g[t]       + b[t]       + r0;
    xn[n * 512 + t + 256] = d1 * inv * g[t + 256] + b[t + 256] + r1;
}

__global__ void k_poolsum(const float* __restrict__ x, const int* __restrict__ batch,
                          float* __restrict__ sums, float* __restrict__ cnt) {
    int i = blockIdx.x * blockDim.x + threadIdx.x;
    if (i >= NN * 512) return;
    int n = i >> 9, c = i & 511;
    atomicAdd(&sums[batch[n] * 512 + c], x[i]);
    if (c == 0) atomicAdd(&cnt[batch[n]], 1.0f);
}

__global__ void k_poolfin(const float* __restrict__ sums, const float* __restrict__ cnt,
                          float* __restrict__ go) {
    int i = blockIdx.x * blockDim.x + threadIdx.x;
    if (i < NG * 512) go[i] = sums[i] / fmaxf(cnt[i >> 9], 1.0f);
}

// ---------- launch ----------
extern "C" void kernel_launch(void* const* d_in, const int* in_sizes, int n_in,
                              void* d_out, int out_size, void* d_ws, size_t ws_size,
                              hipStream_t stream) {
    const float* x     = (const float*)d_in[0];
    const int*   ei    = (const int*)  d_in[1];
    const int*   nt    = (const int*)  d_in[2];
    const int*   batch = (const int*)  d_in[3];
    const float* emb   = (const float*)d_in[4];
    const float* W[3]    = {(const float*)d_in[5],  (const float*)d_in[9],  (const float*)d_in[13]};
    const float* asrc[3] = {(const float*)d_in[6],  (const float*)d_in[10], (const float*)d_in[14]};
    const float* adst[3] = {(const float*)d_in[7],  (const float*)d_in[11], (const float*)d_in[15]};
    const float* bias[3] = {(const float*)d_in[8],  (const float*)d_in[12], (const float*)d_in[16]};
    const float* lng   = (const float*)d_in[17];   // [3][512]
    const float* lnb   = (const float*)d_in[18];
    float* out = (float*)d_out;                    // [10000*512] node emb + [8*512] graph emb

    // workspace carve-up (~87 MB total)
    char* p = (char*)d_ws;
    _Float16* Wt  = (_Float16*)p; p += 3 * 512 * 512 * 2;
    float* XA   = (float*)p; p += (size_t)NN * 512 * 4;
    float* XB   = (float*)p; p += (size_t)NN * 512 * 4;
    float* Hb   = (float*)p; p += (size_t)NN * 512 * 4;
    float* OUTb = (float*)p; p += (size_t)NN * 512 * 4;
    float* ALS  = (float*)p; p += NN * 4 * 4;
    float* ALD  = (float*)p; p += NN * 4 * 4;
    unsigned* MX = (unsigned*)p; p += NN * 4 * 4;
    float* SDEN = (float*)p; p += NN * 4 * 4;
    float* EW   = (float*)p; p += (size_t)ET * 4 * 4;
    float* SUMS = (float*)p; p += NG * 512 * 4;
    float* CNT  = (float*)p; p += 64;

    // weights -> f16 transposed [N][K]
    for (int l = 0; l < 3; l++)
        k_wt<<<1024, 256, 0, stream>>>(W[l], Wt + (size_t)l * 512 * 512);

    // x + type_emb
    k_addemb<<<(NN * 512 + 255) / 256, 256, 0, stream>>>(x, nt, emb, XA);

    float* Xcur = XA;
    for (int l = 0; l < 3; l++) {
        const int Hh     = (l < 2) ? 4 : 1;
        const int cshift = (l < 2) ? 7 : 9;      // C = 128 or 512

        k_gemm<<<313, 256, 0, stream>>>(Xcur, Wt + (size_t)l * 512 * 512, Hb);
        k_alpha<<<1250, 256, 0, stream>>>(Hb, asrc[l], adst[l], ALS, ALD, Hh, cshift);

        k_zero<<<(NN * Hh + 255) / 256, 256, 0, stream>>>((float*)MX, NN * Hh);
        k_zero<<<(NN * Hh + 255) / 256, 256, 0, stream>>>(SDEN, NN * Hh);
        k_zero<<<(NN * 512 + 255) / 256, 256, 0, stream>>>(OUTb, NN * 512);

        k_edge1<<<(ET * Hh + 255) / 256, 256, 0, stream>>>(ei, ALS, ALD, EW, MX, Hh);
        k_edge2<<<(ET * Hh + 255) / 256, 256, 0, stream>>>(ei, MX, EW, SDEN, Hh);
        k_edge3<<<(ET * 128 + 255) / 256, 256, 0, stream>>>(ei, Hb, EW, SDEN, OUTb, Hh, cshift);

        float* xnext = (l == 2) ? out : ((Xcur == XA) ? XB : XA);
        k_post<<<NN, 256, 0, stream>>>(OUTb, bias[l], lng + l * 512, lnb + l * 512,
                                       (l == 0) ? nullptr : Xcur, xnext, (l < 2) ? 1 : 0);
        Xcur = xnext;
    }

    // graph mean pool (reads node embeddings already written into d_out)
    k_zero<<<(NG * 512 + 255) / 256, 256, 0, stream>>>(SUMS, NG * 512);
    k_zero<<<1, 64, 0, stream>>>(CNT, NG);
    k_poolsum<<<(NN * 512 + 255) / 256, 256, 0, stream>>>(out, batch, SUMS, CNT);
    k_poolfin<<<(NG * 512 + 255) / 256, 256, 0, stream>>>(SUMS, CNT, out + (size_t)NN * 512);
}
